// MusicTransformer_79370995630131
// MI455X (gfx1250) — compile-verified, matrix operand order checked by reading, not run on hardware
//
#include <hip/hip_runtime.h>

// ---------------------------------------------------------------------------
// Music Transformer forward, CDNA5 (gfx1250).
// All matmuls run on v_wmma_f32_16x16x32_bf16 with bf16 operands resident in
// memory: every A/B fragment is exactly two global_load_b128 (the ISA fragment
// layout is two contiguous 8-element K-runs per lane), zero conversions and
// zero divergence in the inner loop.  Wave tile = 32x64 (2 A frags x 4 B
// frags -> 8 WMMAs per 12 b128 loads).  fp32 is kept only where the math
// needs it: residual stream, scores pre-softmax, LN/softmax stats, logits.
// ---------------------------------------------------------------------------

typedef __bf16 v16bf  __attribute__((ext_vector_type(16)));
typedef __bf16 bf16x8 __attribute__((ext_vector_type(8)));
typedef float  v8f    __attribute__((ext_vector_type(8)));

#define BB      4
#define SS      512
#define DD      512
#define HH      8
#define HDSZ    64
#define DFFSZ   1024
#define NLAYER  6
#define NVOCAB  390
#define MAXSEQP 2048
#define BSROWS  (BB * SS)

// fp32 -> bf16 round-to-nearest-even (integer path, no cvt dependency)
__device__ __forceinline__ __bf16 f2bf(float f) {
    unsigned u = __builtin_bit_cast(unsigned, f);
    unsigned r = (u + 0x7FFFu + ((u >> 16) & 1u)) >> 16;
    unsigned short h = (unsigned short)r;
    return __builtin_bit_cast(__bf16, h);
}

__device__ __forceinline__ v16bf load_frag(const __bf16* p) {
    // ISA 16-bit fragment: elements 0..7 = K[base..base+7], 8..15 = K[base+16..+23]
    bf16x8 lo = *(const bf16x8*)(p);
    bf16x8 hi = *(const bf16x8*)(p + 16);
    return __builtin_shufflevector(lo, hi, 0,1,2,3,4,5,6,7,8,9,10,11,12,13,14,15);
}

// ---------------------------------------------------------------------------
// Batched WMMA GEMM, bf16 in / fp32 accum:  C[z] = alpha * A[z] * W[z]^T + bias
// A: MxK row-major bf16 (lda).  B ("W"): NxK row-major bf16 (ldb) -> K runs
// contiguous, so both operands use b128 fragment loads.
// Wave: 32x64 tile (8 accumulators; 2 shared A frags, 4 shared B frags);
// block = 8 waves = 256x64.  z = (b,h) with independent strides.
// Epilogue modes: 0 = fp32 C;  1 = bf16 C;  2 = QKV pack (q,k -> qkb row-major,
// v -> per-head transposed vT so P.V becomes another W-layout GEMM).
// ---------------------------------------------------------------------------
__global__ __launch_bounds__(256)
void wmma_gemm_kernel(const __bf16* __restrict__ A,  int lda, long sAB, long sAH,
                      const __bf16* __restrict__ Bw, int ldb, long sBB, long sBH,
                      const float* __restrict__ bias,
                      float* __restrict__ Cf, __bf16* __restrict__ Cb,
                      int ldc, long sCB, long sCH,
                      __bf16* __restrict__ qkb, __bf16* __restrict__ vtb,
                      int M, int N, int K, int Hdim,
                      float alpha, int relu, int mode, int nbase)
{
    int z  = blockIdx.z;
    int bi = z / Hdim, hi = z % Hdim;
    A  += (long)bi * sAB + (long)hi * sAH;
    Bw += (long)bi * sBB + (long)hi * sBH;

    int wave = threadIdx.x >> 5;
    int lane = threadIdx.x & 31;
    int m0 = blockIdx.y * 256 + wave * 32;    // wave-uniform
    int n0 = blockIdx.x * 64;
    if (m0 >= M) return;

    int mr   = lane & 15;
    int half = lane >> 4;

    const __bf16* Arow0 = A + (long)(m0 + mr) * lda + half * 8;
    const __bf16* Arow1 = A + (long)(m0 + 16 + mr) * lda + half * 8;

    int nc[4];
    const __bf16* Brow[4];
#pragma unroll
    for (int j = 0; j < 4; ++j) {
        nc[j] = n0 + j * 16 + mr;
        int safe = nc[j] < N ? nc[j] : 0;       // clamp, never predicate loads
        Brow[j] = Bw + (long)safe * ldb + half * 8;
    }

    v8f acc[2][4];
#pragma unroll
    for (int i = 0; i < 2; ++i)
#pragma unroll
        for (int j = 0; j < 4; ++j)
            acc[i][j] = (v8f){0.f,0.f,0.f,0.f,0.f,0.f,0.f,0.f};

    for (int k0 = 0; k0 < K; k0 += 32) {
        v16bf a0 = load_frag(Arow0 + k0);
        v16bf a1 = load_frag(Arow1 + k0);
#pragma unroll
        for (int j = 0; j < 4; ++j) {
            v16bf b = load_frag(Brow[j] + k0);
            acc[0][j] = __builtin_amdgcn_wmma_f32_16x16x32_bf16(
                            false, a0, false, b, (short)0, acc[0][j], false, false);
            acc[1][j] = __builtin_amdgcn_wmma_f32_16x16x32_bf16(
                            false, a1, false, b, (short)0, acc[1][j], false, false);
        }
    }

    long cbase = (long)bi * sCB + (long)hi * sCH;
#pragma unroll
    for (int j = 0; j < 4; ++j) {
        if (nc[j] >= N) continue;
        float bv = bias ? bias[nc[j]] : 0.f;
#pragma unroll
        for (int i = 0; i < 2; ++i) {
#pragma unroll
            for (int r = 0; r < 8; ++r) {
                float o = acc[i][j][r] * alpha + bv;
                if (relu) o = fmaxf(o, 0.f);
                int m = m0 + i * 16 + r + 8 * half;   // C/D layout row
                if (mode == 0) {
                    Cf[cbase + (long)m * ldc + nc[j]] = o;
                } else if (mode == 1) {
                    Cb[cbase + (long)m * ldc + nc[j]] = f2bf(o);
                } else {                               // QKV pack
                    int ng = nbase + nc[j];            // global col in [0, 3D)
                    if (ng < 2 * DD) {
                        qkb[(long)m * (2 * DD) + ng] = f2bf(o);
                    } else {
                        int nl = ng - 2 * DD;          // v-local col
                        int h  = nl >> 6, d = nl & (HDSZ - 1);
                        int b  = m >> 9,  s = m & (SS - 1);
                        vtb[((((long)b * HH + h) * HDSZ + d) * SS) + s] = f2bf(o);
                    }
                }
            }
        }
    }
}

// ---------------------------------------------------------------------------
// Softmax over rows of (z,S,S) fp32 scores -> bf16 probabilities.
// mode 0: plain  1: + skew(qe): s[q,k] += (k<=q)? qe[q,S-1-q+k] : 0
// mode 2: causal.  S == 512.
// ---------------------------------------------------------------------------
__global__ __launch_bounds__(256)
void softmax_kernel(const float* __restrict__ sc, const float* __restrict__ qe,
                    __bf16* __restrict__ prob, int mode)
{
    __shared__ float red[256];
    int z = blockIdx.y, q = blockIdx.x;
    const float* row  = sc + ((long)z * SS + q) * SS;
    const float* qrow = (mode == 1) ? (qe + ((long)z * SS + q) * SS) : nullptr;
    __bf16* prow = prob + ((long)z * SS + q) * SS;
    int tid = threadIdx.x;
    int k0 = tid, k1 = tid + 256;
    float v0 = row[k0], v1 = row[k1];
    if (mode == 1) {
        if (k0 <= q) v0 += qrow[SS - 1 - q + k0];
        if (k1 <= q) v1 += qrow[SS - 1 - q + k1];
    } else if (mode == 2) {
        if (k0 > q) v0 = -__builtin_inff();
        if (k1 > q) v1 = -__builtin_inff();
    }
    red[tid] = fmaxf(v0, v1); __syncthreads();
    for (int o = 128; o > 0; o >>= 1) {
        if (tid < o) red[tid] = fmaxf(red[tid], red[tid + o]);
        __syncthreads();
    }
    float mx = red[0]; __syncthreads();
    float e0 = __expf(v0 - mx), e1 = __expf(v1 - mx);
    red[tid] = e0 + e1; __syncthreads();
    for (int o = 128; o > 0; o >>= 1) {
        if (tid < o) red[tid] += red[tid + o];
        __syncthreads();
    }
    float inv = 1.f / red[0];
    prow[k0] = f2bf(e0 * inv);
    prow[k1] = f2bf(e1 * inv);
}

// ---------------------------------------------------------------------------
// LayerNorm over D=512: outf/outb = LN(x + res) * g + b  (fp32 + bf16 mirror)
// ---------------------------------------------------------------------------
__global__ __launch_bounds__(256)
void layernorm_kernel(const float* __restrict__ x, const float* __restrict__ res,
                      const float* __restrict__ g, const float* __restrict__ b,
                      float* __restrict__ outf, __bf16* __restrict__ outb)
{
    __shared__ float red[256];
    long row = blockIdx.x;
    int tid = threadIdx.x;
    const float* xr = x + row * DD;
    float a0 = xr[tid], a1 = xr[tid + 256];
    if (res) {
        const float* rr = res + row * DD;
        a0 += rr[tid]; a1 += rr[tid + 256];
    }
    red[tid] = a0 + a1; __syncthreads();
    for (int o = 128; o > 0; o >>= 1) { if (tid < o) red[tid] += red[tid + o]; __syncthreads(); }
    float mu = red[0] * (1.f / DD); __syncthreads();
    float d0 = a0 - mu, d1 = a1 - mu;
    red[tid] = d0 * d0 + d1 * d1; __syncthreads();
    for (int o = 128; o > 0; o >>= 1) { if (tid < o) red[tid] += red[tid + o]; __syncthreads(); }
    float rstd = rsqrtf(red[0] * (1.f / DD) + 1e-5f);
    float y0 = d0 * rstd * g[tid]       + b[tid];
    float y1 = d1 * rstd * g[tid + 256] + b[tid + 256];
    outf[row * DD + tid]       = y0;
    outf[row * DD + tid + 256] = y1;
    outb[row * DD + tid]       = f2bf(y0);
    outb[row * DD + tid + 256] = f2bf(y1);
}

// ---------------------------------------------------------------------------
// Embedding + sinusoidal PE (PE is a host constant -> recomputed on device).
// ---------------------------------------------------------------------------
__global__ __launch_bounds__(256)
void embed_kernel(const int* __restrict__ tok, const float* __restrict__ emb,
                  float* __restrict__ outf, __bf16* __restrict__ outb)
{
    int row = blockIdx.x;                 // b*S + s
    int s   = row % SS;
    int t   = tok[row];
    const float* er = emb + (long)t * DD;
    int tid = threadIdx.x;
#pragma unroll
    for (int j = 0; j < 2; ++j) {
        int d = tid + j * 256;
        float di  = (float)(d & ~1);
        float div = __expf(di * (-9.210340371976184f / (float)DD));
        float ang = (float)s * div;
        float pe  = (d & 1) ? __cosf(ang) : __sinf(ang);
        float v   = er[d] + pe;
        outf[(long)row * DD + d] = v;
        outb[(long)row * DD + d] = f2bf(v);
    }
}

// fp32 -> bf16 bulk conversion (weights, once per launch)
__global__ __launch_bounds__(256)
void cvt_kernel(const float* __restrict__ src, __bf16* __restrict__ dst, long n)
{
    long i = ((long)blockIdx.x * 256 + threadIdx.x) * 4;
    if (i + 3 < n) {
        float4 v = *(const float4*)(src + i);
        dst[i]     = f2bf(v.x); dst[i + 1] = f2bf(v.y);
        dst[i + 2] = f2bf(v.z); dst[i + 3] = f2bf(v.w);
    } else {
        for (long j = i; j < n; ++j) dst[j] = f2bf(src[j]);
    }
}

// ---------------------------------------------------------------------------
// Host orchestration
// ---------------------------------------------------------------------------
struct AttnP { const float *b_in, *b_out, *w_in, *w_out; };
struct FfnP  { const float *b1, *b2, *w1, *w2; };
struct LnP   { const float *b, *s; };
struct AttnB { const __bf16 *w_in, *w_out; };
struct FfnB  { const __bf16 *w1, *w2; };
struct EncL  { AttnP attn; const float* er; FfnP ffn; LnP ln1, ln2;
               AttnB attnb; const __bf16* erb; FfnB ffnb; };
struct DecL  { AttnP cross1, cross2; FfnP ffn; LnP ln1, ln2, ln3, ln4; AttnP self_attn;
               AttnB cross1b, cross2b, selfb; FfnB ffnb; };

extern "C" void kernel_launch(void* const* d_in, const int* in_sizes, int n_in,
                              void* d_out, int out_size, void* d_ws, size_t ws_size,
                              hipStream_t stream) {
    (void)in_sizes; (void)n_in; (void)out_size; (void)ws_size;

    const int* x1  = (const int*)d_in[0];
    const int* x2  = (const int*)d_in[1];
    const int* tgt = (const int*)d_in[2];

    // Params: JAX pytree leaves, dicts flattened in SORTED key order.
    int p = 3;
    auto nf = [&]() { return (const float*)d_in[p++]; };
    auto rdAttn = [&](AttnP& a) { a.b_in = nf(); a.b_out = nf(); a.w_in = nf(); a.w_out = nf(); };
    auto rdFfn  = [&](FfnP& f)  { f.b1 = nf(); f.b2 = nf(); f.w1 = nf(); f.w2 = nf(); };
    auto rdLn   = [&](LnP& l)   { l.b = nf(); l.s = nf(); };

    DecL dec[NLAYER];
    for (int i = 0; i < NLAYER; ++i) {   // cross1,cross2,ffn,ln1..ln4,self_attn
        rdAttn(dec[i].cross1); rdAttn(dec[i].cross2); rdFfn(dec[i].ffn);
        rdLn(dec[i].ln1); rdLn(dec[i].ln2); rdLn(dec[i].ln3); rdLn(dec[i].ln4);
        rdAttn(dec[i].self_attn);
    }
    const float* embW = nf();
    EncL enc1[NLAYER], enc2[NLAYER];
    for (int i = 0; i < NLAYER; ++i) {   // attn,er,ffn,ln1,ln2
        rdAttn(enc1[i].attn); enc1[i].er = nf(); rdFfn(enc1[i].ffn);
        rdLn(enc1[i].ln1); rdLn(enc1[i].ln2);
    }
    for (int i = 0; i < NLAYER; ++i) {
        rdAttn(enc2[i].attn); enc2[i].er = nf(); rdFfn(enc2[i].ffn);
        rdLn(enc2[i].ln1); rdLn(enc2[i].ln2);
    }
    LnP final_ln; rdLn(final_ln);
    const float* wout_b = nf();
    const float* wout_w = nf();

    // ---- workspace -------------------------------------------------------
    const long X  = (long)BSROWS * DD;
    const long SC = (long)BB * HH * SS * SS;
    float* fp   = (float*)d_ws;
    float* cur    = fp;  fp += X;          // residual stream (fp32)
    float* t_gemm = fp;  fp += X;          // sublayer output pre-LN (fp32)
    float* scores = fp;  fp += SC;         // (B,H,S,S)
    float* qe     = fp;  fp += SC;

    __bf16* bp = (__bf16*)fp;
    auto balloc = [&](long n) { __bf16* q = bp; bp += n; return q; };
    __bf16* curb   = balloc(X);            // bf16 mirror of residual stream
    __bf16* m1b    = balloc(X);            // encoder memories (bf16, GEMM A)
    __bf16* m2b    = balloc(X);
    __bf16* qkb    = balloc(2 * X);        // packed q,k  (B,S,2D)
    __bf16* vtb    = balloc(X);            // V transposed (B,H,HD,S)
    __bf16* tattnb = balloc(X);            // merged heads (bf16)
    __bf16* tffnb  = balloc(2 * X);        // relu(FFN1)   (B,S,DFF)
    __bf16* probb  = balloc(SC);           // softmax probabilities

    auto cvt = [&](const float* src, long n) {
        __bf16* dst = balloc(n);
        cvt_kernel<<<dim3((unsigned)((n + 1023) / 1024)), 256, 0, stream>>>(src, dst, n);
        return (const __bf16*)dst;
    };
    // weight arena (bf16), converted once per launch
    for (int i = 0; i < NLAYER; ++i) {
        dec[i].selfb   = { cvt(dec[i].self_attn.w_in, 3L*DD*DD), cvt(dec[i].self_attn.w_out, (long)DD*DD) };
        dec[i].cross1b = { cvt(dec[i].cross1.w_in,    3L*DD*DD), cvt(dec[i].cross1.w_out,    (long)DD*DD) };
        dec[i].cross2b = { cvt(dec[i].cross2.w_in,    3L*DD*DD), cvt(dec[i].cross2.w_out,    (long)DD*DD) };
        dec[i].ffnb    = { cvt(dec[i].ffn.w1, (long)DFFSZ*DD), cvt(dec[i].ffn.w2, (long)DD*DFFSZ) };
    }
    for (int i = 0; i < NLAYER; ++i) {
        enc1[i].attnb = { cvt(enc1[i].attn.w_in, 3L*DD*DD), cvt(enc1[i].attn.w_out, (long)DD*DD) };
        enc1[i].erb   = cvt(enc1[i].er, (long)MAXSEQP * HDSZ);
        enc1[i].ffnb  = { cvt(enc1[i].ffn.w1, (long)DFFSZ*DD), cvt(enc1[i].ffn.w2, (long)DD*DFFSZ) };
        enc2[i].attnb = { cvt(enc2[i].attn.w_in, 3L*DD*DD), cvt(enc2[i].attn.w_out, (long)DD*DD) };
        enc2[i].erb   = cvt(enc2[i].er, (long)MAXSEQP * HDSZ);
        enc2[i].ffnb  = { cvt(enc2[i].ffn.w1, (long)DFFSZ*DD), cvt(enc2[i].ffn.w2, (long)DD*DFFSZ) };
    }
    const __bf16* woutb = cvt(wout_w, (long)NVOCAB * DD);

    // ---- launch helpers --------------------------------------------------
    auto gemm = [&](const __bf16* Ap, int lda, long sAB, long sAH,
                    const __bf16* Bp, int ldb, long sBB, long sBH,
                    const float* bias, float* Cfp, __bf16* Cbp,
                    int ldc, long sCB, long sCH,
                    __bf16* qk, __bf16* vt,
                    int M, int N, int K, int Hdim, int batch,
                    float alpha, int relu, int mode, int nbase) {
        dim3 grid((N + 63) / 64, (M + 255) / 256, batch);
        wmma_gemm_kernel<<<grid, 256, 0, stream>>>(
            Ap, lda, sAB, sAH, Bp, ldb, sBB, sBH, bias,
            Cfp, Cbp, ldc, sCB, sCH, qk, vt,
            M, N, K, Hdim, alpha, relu, mode, nbase);
    };
    auto lnorm = [&](const float* x, const float* res, LnP l,
                     float* outf, __bf16* outb) {
        layernorm_kernel<<<BSROWS, 256, 0, stream>>>(x, res, l.s, l.b, outf, outb);
    };

    const long sQB = (long)SS * 2 * DD;    // qkb per-batch stride
    const long sVB = (long)HH * HDSZ * SS; // vtb per-batch stride
    const long sSB = (long)HH * SS * SS;
    const long sSH = (long)SS * SS;

    // xkvb==nullptr => self-attention.  mode: 0 plain, 1 relative, 2 causal.
    auto attention = [&](const __bf16* xqb, const __bf16* xkvb,
                         const AttnP& w, const AttnB& wb,
                         const __bf16* erb, int mode, float* outp) {
        if (!xkvb) {
            gemm(xqb, DD, 0, 0, wb.w_in, DD, 0, 0, w.b_in,
                 nullptr, nullptr, 0, 0, 0, qkb, vtb,
                 BSROWS, 3 * DD, DD, 1, 1, 1.f, 0, 2, 0);
        } else {
            gemm(xqb, DD, 0, 0, wb.w_in, DD, 0, 0, w.b_in,
                 nullptr, nullptr, 0, 0, 0, qkb, vtb,
                 BSROWS, DD, DD, 1, 1, 1.f, 0, 2, 0);
            gemm(xkvb, DD, 0, 0, wb.w_in + (long)DD * DD, DD, 0, 0, w.b_in + DD,
                 nullptr, nullptr, 0, 0, 0, qkb, vtb,
                 BSROWS, 2 * DD, DD, 1, 1, 1.f, 0, 2, DD);
        }
        // scores = (q/8) . k^T per (b,h):  A = q rows, W = k rows (both in qkb)
        gemm(qkb, 2 * DD, sQB, HDSZ, qkb + DD, 2 * DD, sQB, HDSZ, nullptr,
             scores, nullptr, SS, sSB, sSH, nullptr, nullptr,
             SS, SS, HDSZ, HH, BB * HH, 0.125f, 0, 0, 0);
        if (mode == 1)
            gemm(qkb, 2 * DD, sQB, HDSZ,
                 erb + (long)(MAXSEQP - SS) * HDSZ, HDSZ, 0, 0, nullptr,
                 qe, nullptr, SS, sSB, sSH, nullptr, nullptr,
                 SS, SS, HDSZ, HH, BB * HH, 0.125f, 0, 0, 0);
        softmax_kernel<<<dim3(SS, BB * HH), 256, 0, stream>>>(scores, qe, probb, mode);
        // head_out = P . V:  W = vT (HD x S rows, K contiguous)
        gemm(probb, SS, sSB, sSH, vtb, SS, sVB, (long)HDSZ * SS, nullptr,
             nullptr, tattnb, DD, (long)SS * DD, HDSZ, nullptr, nullptr,
             SS, HDSZ, SS, HH, BB * HH, 1.f, 0, 1, 0);
        gemm(tattnb, DD, 0, 0, wb.w_out, DD, 0, 0, w.b_out,
             outp, nullptr, DD, 0, 0, nullptr, nullptr,
             BSROWS, DD, DD, 1, 1, 1.f, 0, 0, 0);
    };
    auto ffn = [&](const __bf16* xb, const FfnP& f, const FfnB& fb, float* outp) {
        gemm(xb, DD, 0, 0, fb.w1, DD, 0, 0, f.b1,
             nullptr, tffnb, DFFSZ, 0, 0, nullptr, nullptr,
             BSROWS, DFFSZ, DD, 1, 1, 1.f, 1, 1, 0);
        gemm(tffnb, DFFSZ, 0, 0, fb.w2, DFFSZ, 0, 0, f.b2,
             outp, nullptr, DD, 0, 0, nullptr, nullptr,
             BSROWS, DD, DFFSZ, 1, 1, 1.f, 0, 0, 0);
    };
    auto encoder = [&](const int* toks, EncL* layers, __bf16* memb) {
        embed_kernel<<<BSROWS, 256, 0, stream>>>(toks, embW, cur, curb);
        for (int i = 0; i < NLAYER; ++i) {
            attention(curb, nullptr, layers[i].attn, layers[i].attnb,
                      layers[i].erb, 1, t_gemm);
            lnorm(t_gemm, cur, layers[i].ln1, cur, curb);
            ffn(curb, layers[i].ffn, layers[i].ffnb, t_gemm);
            lnorm(t_gemm, cur, layers[i].ln2, cur, curb);
        }
        lnorm(cur, nullptr, final_ln, t_gemm, memb);  // fp32 side is scratch
    };

    encoder(x1, enc1, m1b);
    encoder(x2, enc2, m2b);

    // Decoder
    embed_kernel<<<BSROWS, 256, 0, stream>>>(tgt, embW, cur, curb);
    for (int i = 0; i < NLAYER; ++i) {
        attention(curb, nullptr, dec[i].self_attn, dec[i].selfb, nullptr, 2, t_gemm);
        lnorm(t_gemm, cur, dec[i].ln1, cur, curb);
        attention(curb, m1b, dec[i].cross1, dec[i].cross1b, nullptr, 0, t_gemm);
        lnorm(t_gemm, cur, dec[i].ln2, cur, curb);
        attention(curb, m2b, dec[i].cross2, dec[i].cross2b, nullptr, 0, t_gemm);
        lnorm(t_gemm, cur, dec[i].ln3, cur, curb);
        ffn(curb, dec[i].ffn, dec[i].ffnb, t_gemm);
        lnorm(t_gemm, cur, dec[i].ln4, cur, curb);
    }
    lnorm(cur, nullptr, final_ln, t_gemm, curb);

    // Vocab head -> d_out fp32 (B,S,VOCAB)
    gemm(curb, DD, 0, 0, woutb, DD, 0, 0, wout_b,
         (float*)d_out, nullptr, NVOCAB, 0, 0, nullptr, nullptr,
         BSROWS, NVOCAB, DD, 1, 1, 1.f, 0, 0, 0);
}
